// SimpleMambaLayer_62663572848655
// MI455X (gfx1250) — compile-verified, matrix operand order checked by reading, not run on hardware
//
#include <hip/hip_runtime.h>
#include <hip/hip_bf16.h>

#define D_MODEL 768
#define D_INNER 768
#define NPROJ   (2*D_INNER)   // 1536
#define BATCH   4
#define SEQ     8192
#define MTOT    (BATCH*SEQ)   // 32768
#define CHUNK   128
#define NCHUNK  (SEQ/CHUNK)   // 64

typedef _Float16 v8h  __attribute__((ext_vector_type(8)));
typedef _Float16 v16h __attribute__((ext_vector_type(16)));
typedef float    v8f  __attribute__((ext_vector_type(8)));

__device__ __forceinline__ float silu_f(float v) { return v / (1.0f + __expf(-v)); }

// ---------------- prep kernels ----------------
__global__ void k_cvt_f16(const float* __restrict__ src, _Float16* __restrict__ dst, int n) {
  int i = blockIdx.x * blockDim.x + threadIdx.x;
  int stride = gridDim.x * blockDim.x;
  for (; i < n; i += stride) dst[i] = (_Float16)src[i];
}

// src [K][N] row-major f32  ->  dst [N][K] row-major f16
__global__ void k_transpose_cvt(const float* __restrict__ src, _Float16* __restrict__ dst,
                                int K, int N) {
  int i = blockIdx.x * blockDim.x + threadIdx.x;
  int stride = gridDim.x * blockDim.x;
  int tot = K * N;
  for (; i < tot; i += stride) {
    int k = i / N, n = i % N;
    dst[(size_t)n * K + k] = (_Float16)src[i];
  }
}

// ---------------- WMMA GEMM ----------------
// C[M,N] = A[M,K] * B[K,N],  A row-major [M,K] f16, Bt row-major [N,K] f16.
// Block: 256 threads = 8 wave32, 128x128 tile. Wave grid 4(M) x 2(N); each
// wave owns a 32x64 sub-tile = 2x4 WMMA 16x16 tiles. K-step 32, LDS double
// buffered, padded stride 40 halves (80B) -> conflict-free b128 frag loads.
template<int N_DIM, int K_DIM, bool HALF_OUT>
__global__ __launch_bounds__(256)
void k_gemm_wmma(const _Float16* __restrict__ A,
                 const _Float16* __restrict__ Bt,
                 void* __restrict__ Cout) {
  constexpr int KSTEPS = K_DIM / 32;
  __shared__ __align__(16) _Float16 As[2][128][40];
  __shared__ __align__(16) _Float16 Bs[2][128][40];

  const int tid  = threadIdx.x;
  const int lane = tid & 31;
  const int wid  = tid >> 5;
  const int wM   = wid & 3;   // 0..3
  const int wN   = wid >> 2;  // 0..1
  const int tM0  = blockIdx.x * 128;
  const int tN0  = blockIdx.y * 128;

  // tile loader mapping: each thread does 2 b128 loads per operand per k-step
  const int ldRow = tid >> 2;          // 0..63
  const int ldCol = (tid & 3) << 3;    // 0,8,16,24

  const _Float16* gA = A  + (size_t)tM0 * K_DIM;
  const _Float16* gB = Bt + (size_t)tN0 * K_DIM;

  v8h sa[2], sb[2];
  #pragma unroll
  for (int i = 0; i < 2; i++) {
    sa[i] = *(const v8h*)(gA + (size_t)(ldRow + 64 * i) * K_DIM + ldCol);
    sb[i] = *(const v8h*)(gB + (size_t)(ldRow + 64 * i) * K_DIM + ldCol);
  }
  #pragma unroll
  for (int i = 0; i < 2; i++) {
    *(v8h*)&As[0][ldRow + 64 * i][ldCol] = sa[i];
    *(v8h*)&Bs[0][ldRow + 64 * i][ldCol] = sb[i];
  }
  __syncthreads();

  v8f acc[2][4] = {};

  const int kb  = (lane >> 4) << 3;  // 0 or 8   (ISA 16-bit A/B frag layout)
  const int rsl = lane & 15;

  for (int ks = 0; ks < KSTEPS; ks++) {
    const int buf = ks & 1;
    if (ks + 1 < KSTEPS) {
      const int ko = (ks + 1) * 32;
      #pragma unroll
      for (int i = 0; i < 2; i++) {
        sa[i] = *(const v8h*)(gA + (size_t)(ldRow + 64 * i) * K_DIM + ko + ldCol);
        sb[i] = *(const v8h*)(gB + (size_t)(ldRow + 64 * i) * K_DIM + ko + ldCol);
      }
    }

    v16h af[2], bf[4];
    #pragma unroll
    for (int t = 0; t < 2; t++) {
      const int r = wM * 32 + t * 16 + rsl;
      v8h lo = *(const v8h*)&As[buf][r][kb];
      v8h hi = *(const v8h*)&As[buf][r][kb + 16];
      af[t] = __builtin_shufflevector(lo, hi, 0,1,2,3,4,5,6,7,8,9,10,11,12,13,14,15);
    }
    #pragma unroll
    for (int t = 0; t < 4; t++) {
      const int r = wN * 64 + t * 16 + rsl;
      v8h lo = *(const v8h*)&Bs[buf][r][kb];
      v8h hi = *(const v8h*)&Bs[buf][r][kb + 16];
      bf[t] = __builtin_shufflevector(lo, hi, 0,1,2,3,4,5,6,7,8,9,10,11,12,13,14,15);
    }

    #pragma unroll
    for (int a = 0; a < 2; a++)
      #pragma unroll
      for (int b = 0; b < 4; b++)
        acc[a][b] = __builtin_amdgcn_wmma_f32_16x16x32_f16(
            false, af[a], false, bf[b], (short)0, acc[a][b], false, false);

    if (ks + 1 < KSTEPS) {
      const int nb = buf ^ 1;
      #pragma unroll
      for (int i = 0; i < 2; i++) {
        *(v8h*)&As[nb][ldRow + 64 * i][ldCol] = sa[i];
        *(v8h*)&Bs[nb][ldRow + 64 * i][ldCol] = sb[i];
      }
    }
    __syncthreads();
  }

  // epilogue: C/D layout — VGPR r: M = r + 8*(lane>=16); N = lane&15
  const int rhi = (lane >> 4) << 3;
  #pragma unroll
  for (int a = 0; a < 2; a++) {
    #pragma unroll
    for (int b = 0; b < 4; b++) {
      const size_t rowb = (size_t)(tM0 + wM * 32 + a * 16 + rhi);
      const int    col  = tN0 + wN * 64 + b * 16 + rsl;
      #pragma unroll
      for (int r = 0; r < 8; r++) {
        const size_t off = (rowb + r) * (size_t)N_DIM + col;
        if (HALF_OUT) ((_Float16*)Cout)[off] = (_Float16)acc[a][b][r];
        else          ((float*)Cout)[off]    = acc[a][b][r];
      }
    }
  }
}

// ---------------- chunked scan (conv + silu + cumsum + gate) ----------------
// xz layout: [B][SEQ][1536] f16; xc = cols 0..767, z = cols 768..1535.
__global__ __launch_bounds__(256)
void k_chunk_sums(const _Float16* __restrict__ xz,
                  const float* __restrict__ conv_w,
                  const float* __restrict__ conv_b,
                  float* __restrict__ csum) {
  const int ch = blockIdx.x;            // chunk
  const int c  = blockIdx.y * 256 + threadIdx.x;
  const int b  = blockIdx.z;
  const float w0 = conv_w[c * 3 + 0], w1 = conv_w[c * 3 + 1], w2 = conv_w[c * 3 + 2];
  const float bias = conv_b[c];
  const size_t base = (size_t)b * SEQ * NPROJ + c;
  const int t0 = ch * CHUNK;
  float xm1 = (t0 >= 1) ? (float)xz[base + (size_t)(t0 - 1) * NPROJ] : 0.0f;
  float xm2 = (t0 >= 2) ? (float)xz[base + (size_t)(t0 - 2) * NPROJ] : 0.0f;
  float s = 0.0f;
  for (int t = t0; t < t0 + CHUNK; t++) {
    float xc = (float)xz[base + (size_t)t * NPROJ];
    float cv = w0 * xm2 + w1 * xm1 + w2 * xc + bias;
    s += silu_f(cv);
    xm2 = xm1; xm1 = xc;
  }
  csum[((size_t)b * NCHUNK + ch) * D_INNER + c] = s;
}

__global__ void k_chunk_prefix(float* __restrict__ csum) {
  const int c = blockIdx.x * 256 + threadIdx.x;
  const int b = blockIdx.y;
  float carry = 0.0f;
  for (int ch = 0; ch < NCHUNK; ch++) {
    const size_t idx = ((size_t)b * NCHUNK + ch) * D_INNER + c;
    float v = csum[idx];
    csum[idx] = carry;    // in-place exclusive prefix
    carry += v;
  }
}

__global__ __launch_bounds__(256)
void k_scan_apply(const _Float16* __restrict__ xz,
                  const float* __restrict__ conv_w,
                  const float* __restrict__ conv_b,
                  const float* __restrict__ Bm,
                  const float* __restrict__ Cm,
                  const float* __restrict__ Dv,
                  const float* __restrict__ csum,
                  _Float16* __restrict__ u) {
  const int ch = blockIdx.x;
  const int c  = blockIdx.y * 256 + threadIdx.x;
  const int b  = blockIdx.z;
  const float w0 = conv_w[c * 3 + 0], w1 = conv_w[c * 3 + 1], w2 = conv_w[c * 3 + 2];
  const float bias = conv_b[c];
  float BC = 0.0f;
  #pragma unroll
  for (int s = 0; s < 8; s++) BC += Bm[c * 8 + s] * Cm[c * 8 + s];
  const float Dc = Dv[c];
  const size_t base = (size_t)b * SEQ * NPROJ + c;
  const int t0 = ch * CHUNK;
  float run = csum[((size_t)b * NCHUNK + ch) * D_INNER + c];
  float xm1 = (t0 >= 1) ? (float)xz[base + (size_t)(t0 - 1) * NPROJ] : 0.0f;
  float xm2 = (t0 >= 2) ? (float)xz[base + (size_t)(t0 - 2) * NPROJ] : 0.0f;
  for (int t = t0; t < t0 + CHUNK; t++) {
    float xc = (float)xz[base + (size_t)t * NPROJ];
    float cv = w0 * xm2 + w1 * xm1 + w2 * xc + bias;
    float sv = silu_f(cv);
    run += sv;
    float y = run * BC + sv * Dc;
    float zr = (float)xz[base + (size_t)t * NPROJ + D_INNER];
    u[((size_t)b * SEQ + t) * D_INNER + c] = (_Float16)(y * silu_f(zr));
    xm2 = xm1; xm1 = xc;
  }
}

// ---------------- launcher ----------------
extern "C" void kernel_launch(void* const* d_in, const int* in_sizes, int n_in,
                              void* d_out, int out_size, void* d_ws, size_t ws_size,
                              hipStream_t stream) {
  (void)in_sizes; (void)n_in; (void)out_size; (void)ws_size;
  const float* x      = (const float*)d_in[0];
  const float* W_in   = (const float*)d_in[1];
  const float* conv_w = (const float*)d_in[2];
  const float* conv_b = (const float*)d_in[3];
  const float* Bm     = (const float*)d_in[4];
  const float* Cm     = (const float*)d_in[5];
  const float* Dv     = (const float*)d_in[6];
  const float* W_out  = (const float*)d_in[7];
  float* out = (float*)d_out;

  char* ws = (char*)d_ws;
  _Float16* xz16  = (_Float16*)(ws + 0);          // 32768*1536*2 = 100663296
  _Float16* x16   = (_Float16*)(ws + 100663296);  // 32768*768*2  =  50331648
  _Float16* u16   = (_Float16*)(ws + 150994944);  // 32768*768*2  =  50331648
  _Float16* WinT  = (_Float16*)(ws + 201326592);  // 1536*768*2   =   2359296
  _Float16* WoutT = (_Float16*)(ws + 203685888);  //  768*768*2   =   1179648
  float*    csum  = (float*)   (ws + 204865536);  // 4*64*768*4   =    786432

  k_cvt_f16<<<4096, 256, 0, stream>>>(x, x16, MTOT * D_MODEL);
  k_transpose_cvt<<<2048, 256, 0, stream>>>(W_in,  WinT,  D_MODEL, NPROJ);
  k_transpose_cvt<<<1024, 256, 0, stream>>>(W_out, WoutT, D_INNER, D_MODEL);

  k_gemm_wmma<NPROJ, D_MODEL, true>
      <<<dim3(MTOT / 128, NPROJ / 128), 256, 0, stream>>>(x16, WinT, (void*)xz16);

  k_chunk_sums<<<dim3(NCHUNK, D_INNER / 256, BATCH), 256, 0, stream>>>(xz16, conv_w, conv_b, csum);
  k_chunk_prefix<<<dim3(D_INNER / 256, BATCH), 256, 0, stream>>>(csum);
  k_scan_apply<<<dim3(NCHUNK, D_INNER / 256, BATCH), 256, 0, stream>>>(
      xz16, conv_w, conv_b, Bm, Cm, Dv, csum, u16);

  k_gemm_wmma<D_MODEL, D_INNER, false>
      <<<dim3(MTOT / 128, D_MODEL / 128), 256, 0, stream>>>(u16, WoutT, (void*)out);
}
